// create_ssn_net_26079041421549
// MI455X (gfx1250) — matrix operand order repaired and checked by reading, not compiled, not run on hardware
//
#include <hip/hip_runtime.h>

// ---------------- problem constants (fixed by setup_inputs) ----------------
#define B_   4
#define C_   128
#define H_   256
#define W_   256
#define HW_  (H_ * W_)
#define K_   256           // spixel_h * spixel_w = 16*16
#define NITER 6            // 5 update iters + final assignment pass
#define NEG_  (-1e9f)
#define EPS_  (1e-8f)

#define TILE 64            // pixels per LDS tile
#define NWG  128           // persistent workgroups per batch image

// ---------------- fused-kernel LDS carve (dynamic LDS, 289 KB) -------------
#define OFF_NUM   0                       // K_*C_ f32            = 131072 B
#define OFF_FEAT  (OFF_NUM + 131072)      // K_*C_ f16 (feat)     =  65536 B
#define OFF_DOTS  (OFF_FEAT + 65536)      // TILE*K_ f32          =  65536 B
#define OFF_W     OFF_DOTS                // K_*TILE f16 ALIASES dots (32768 B)
#define OFF_XT    (OFF_DOTS + 65536)      // TILE*C_ f16          =  16384 B
#define OFF_XH    (OFF_XT + 16384)        // C_*TILE f16          =  16384 B
#define OFF_DEN   (OFF_XH + 16384)        // K_ f32               =   1024 B
#define FUSED_LDS (OFF_DEN + 1024)        // 295936 B total (<= 320 KB/WGP)
#define INIT_LDS  (131072 + 1024)

typedef __attribute__((ext_vector_type(16))) _Float16 v16h;
typedef __attribute__((ext_vector_type(8)))  _Float16 v8h;
typedef __attribute__((ext_vector_type(8)))  float    v8f;

// A-matrix fragment (16x32 f16): lanes 0-15 hold K 0..15, lanes 16-31 K 16..31
// (same K order as B).  row = M-row pointer, K contiguous.
__device__ __forceinline__ v16h load_fragA(const _Float16* row, int kc, int hs) {
  return *(const v16h*)(row + kc + 16 * hs);
}
// B-matrix fragment (32x16 f16): lanes 0-15 hold K 0..15, lanes 16-31 K 16..31;
// per-lane the 16 K values are contiguous -> one 32B load from a K-contiguous row.
__device__ __forceinline__ v16h load_fragB(const _Float16* row, int kc, int hs) {
  return *(const v16h*)(row + kc + 16 * hs);
}

// ---------------------------------------------------------------------------
// Init: hard scatter-mean of x into K superpixels.  LDS-aggregated atomics.
// ---------------------------------------------------------------------------
__global__ void __launch_bounds__(256)
init_feat_kernel(const float* __restrict__ x, const int* __restrict__ init_index,
                 float* __restrict__ g_num, float* __restrict__ g_den) {
  extern __shared__ char sm[];
  float* s_num = (float*)sm;             // K_*C_
  float* s_den = (float*)(sm + 131072);  // K_
  const int tid = threadIdx.x;
  for (int i = tid; i < K_ * C_; i += 256) s_num[i] = 0.f;
  for (int i = tid; i < K_; i += 256) s_den[i] = 0.f;
  __syncthreads();

  const int blkPerB = HW_ / 256;
  const int b  = blockIdx.x / blkPerB;
  const int hw = (blockIdx.x % blkPerB) * 256 + tid;
  const int k  = init_index[(size_t)b * HW_ + hw];
  atomicAdd(&s_den[k], 1.f);
  const float* xp = x + (size_t)b * C_ * HW_ + hw;
  float* np = s_num + k * C_;
#pragma unroll 4
  for (int c = 0; c < C_; ++c) atomicAdd(&np[c], xp[(size_t)c * HW_]);
  __syncthreads();

  float* gb = g_num + (size_t)b * K_ * C_;
  for (int i = tid; i < K_ * C_; i += 256) {
    float v = s_num[i];
    if (v != 0.f) atomicAdd(&gb[i], v);
  }
  for (int i = tid; i < K_; i += 256) {
    float v = s_den[i];
    if (v != 0.f) atomicAdd(&g_den[b * K_ + i], v);
  }
}

// ---------------------------------------------------------------------------
// Finalize: feat = num / max(den, dmin); also f16 copy and s2 = ||feat||^2.
// ---------------------------------------------------------------------------
__global__ void __launch_bounds__(256)
finalize_kernel(const float* __restrict__ g_num, const float* __restrict__ g_den,
                float dmin, float* __restrict__ feat_t, _Float16* __restrict__ feat_h,
                float* __restrict__ s2) {
  const int t = blockIdx.x * blockDim.x + threadIdx.x;  // over B_*K_
  if (t >= B_ * K_) return;
  float den = g_den[t];
  den = den > dmin ? den : dmin;
  const float* np = g_num + (size_t)t * C_;
  float* fp = feat_t + (size_t)t * C_;
  _Float16* hp = feat_h + (size_t)t * C_;
  float acc = 0.f;
#pragma unroll 4
  for (int c = 0; c < C_; ++c) {
    float f = np[c] / den;
    fp[c] = f;
    hp[c] = (_Float16)f;
    acc += f * f;
  }
  s2[t] = acc;
}

// ---------------------------------------------------------------------------
// Fused iteration: async-stage feat into LDS, WMMA dots GEMM -> masked 9-way
// softmax -> densified W (aliases dots) -> WMMA accumulation GEMM into the
// per-WG LDS num accumulator, flushed once with atomics.
// ---------------------------------------------------------------------------
__global__ void __launch_bounds__(128)
fused_iter_kernel(const float* __restrict__ x, const int* __restrict__ p2sp,
                  const int* __restrict__ invis, const _Float16* __restrict__ feat_h,
                  const float* __restrict__ s2, float* __restrict__ g_num,
                  float* __restrict__ g_den, int finalPass,
                  float* __restrict__ out_assoc, float* __restrict__ out_lab,
                  int* __restrict__ g_labels) {
  extern __shared__ char sm[];
  float*    sm_num  = (float*)(sm + OFF_NUM);
  _Float16* sm_feat = (_Float16*)(sm + OFF_FEAT);
  float*    sm_dots = (float*)(sm + OFF_DOTS);
  _Float16* sm_w    = (_Float16*)(sm + OFF_W);     // aliases sm_dots region
  _Float16* sm_xt   = (_Float16*)(sm + OFF_XT);
  _Float16* sm_xh   = (_Float16*)(sm + OFF_XH);
  float*    sm_den  = (float*)(sm + OFF_DEN);

  const int tid  = threadIdx.x;
  const int wave = tid >> 5;
  const int lane = tid & 31;    // wave32
  const int Lm   = lane & 15;
  const int hs   = lane >> 4;

  const int b  = blockIdx.x / NWG;
  const int wg = blockIdx.x % NWG;

  const float*    xb  = x + (size_t)b * C_ * HW_;
  const _Float16* fb  = feat_h + (size_t)b * K_ * C_;
  const float*    s2b = s2 + (size_t)b * K_;

  // ---- async DMA the batch's f16 feat matrix into LDS (64 KB, once) ----
  {
    const uint32_t lbase = (uint32_t)(uintptr_t)sm_feat;  // LDS aperture: low 32b = LDS offset
    const char* gsrc = (const char*)fb;
    for (int i = tid; i < (K_ * C_ * 2) / 16; i += 128) {
      uint32_t laddr = lbase + (uint32_t)i * 16u;
      const char* g = gsrc + (size_t)i * 16;
      asm volatile("global_load_async_to_lds_b128 %0, %1, off"
                   :: "v"(laddr), "v"(g) : "memory");
    }
  }
  for (int i = tid; i < K_ * C_; i += 128) sm_num[i] = 0.f;
  for (int i = tid; i < K_; i += 128) sm_den[i] = 0.f;
  asm volatile("s_wait_asynccnt 0" ::: "memory");
  __syncthreads();

  for (int tile = wg; tile < HW_ / TILE; tile += NWG) {
    const int hw0 = tile * TILE;

    // ---- stage x tile f32->f16 in both layouts ----
    for (int i = tid; i < C_ * TILE; i += 128) {
      int c = i / TILE, px = i % TILE;               // coalesced over px
      float v = xb[(size_t)c * HW_ + hw0 + px];
      _Float16 h = (_Float16)v;
      sm_xh[c * TILE + px] = h;                      // [c][px]
      sm_xt[px * C_ + c]   = h;                      // [px][c]
    }
    {  // pull next tile toward L2 (global_prefetch_b8)
      int ntile = tile + NWG;
      if (ntile < HW_ / TILE)
        __builtin_prefetch(xb + (size_t)tid * HW_ + ntile * TILE, 0, 1);
    }
    __syncthreads();

    // ---- GEMM1: dots[sp][px] = feat[sp][c] * x[c][px], all-LDS feeds ----
    {
      const int n = wave;  // each wave owns one 16-pixel column tile
      const _Float16* brow = sm_xt + (n * 16 + Lm) * C_;
      v16h bf0 = load_fragB(brow, 0, hs);
      v16h bf1 = load_fragB(brow, 32, hs);
      v16h bf2 = load_fragB(brow, 64, hs);
      v16h bf3 = load_fragB(brow, 96, hs);
      for (int m = 0; m < 16; ++m) {
        const _Float16* arow = sm_feat + (m * 16 + Lm) * C_;
        v8f acc;
#pragma unroll
        for (int r = 0; r < 8; ++r) acc[r] = 0.f;
        acc = __builtin_amdgcn_wmma_f32_16x16x32_f16(false, load_fragA(arow, 0, hs),
                                                     false, bf0, (short)0, acc, false, false);
        acc = __builtin_amdgcn_wmma_f32_16x16x32_f16(false, load_fragA(arow, 32, hs),
                                                     false, bf1, (short)0, acc, false, false);
        acc = __builtin_amdgcn_wmma_f32_16x16x32_f16(false, load_fragA(arow, 64, hs),
                                                     false, bf2, (short)0, acc, false, false);
        acc = __builtin_amdgcn_wmma_f32_16x16x32_f16(false, load_fragA(arow, 96, hs),
                                                     false, bf3, (short)0, acc, false, false);
        float* dp = sm_dots + (n * 16 + Lm) * K_ + m * 16 + 8 * hs;
#pragma unroll
        for (int r = 0; r < 8; ++r) dp[r] = acc[r];
      }
    }
    __syncthreads();

    // ---- phase A: read dots, masked 9-way softmax into registers ----
    int idxv[9], invv[9], jstar = 0;
    float lg[9];
    if (tid < TILE) {
      const int hw = hw0 + tid;
      float mx = -3.4e38f;
#pragma unroll
      for (int j = 0; j < 9; ++j) {
        int idx = p2sp[((size_t)b * 9 + j) * HW_ + hw];
        int iv  = invis[((size_t)b * 9 + j) * HW_ + hw];
        // softmax(-(x2 - 2d + s2)) == softmax(2d - s2): common -x2 shift cancels
        float l = (iv > 0) ? NEG_ : (2.f * sm_dots[tid * K_ + idx] - s2b[idx]);
        idxv[j] = idx; invv[j] = iv; lg[j] = l;
        mx = l > mx ? l : mx;
      }
      float s = 0.f;
#pragma unroll
      for (int j = 0; j < 9; ++j) { lg[j] = __expf(lg[j] - mx); s += lg[j]; }
      float inv_s = 1.f / s;
      float best = -1.f;
#pragma unroll
      for (int j = 0; j < 9; ++j) {
        float a = lg[j] * inv_s;
        lg[j] = a;
        if (a > best) { best = a; jstar = j; }   // strict > keeps first max
      }
    }
    __syncthreads();   // dots fully consumed; region can be reused as W

    // ---- phase B: zero dense W tile (aliases dots region) ----
    for (int i = tid; i < K_ * TILE; i += 128) sm_w[i] = (_Float16)0.f;
    __syncthreads();

    // ---- phase C: scatter weights into dense W; accumulate den ----
    if (tid < TILE) {
      const int hw = hw0 + tid;
#pragma unroll
      for (int j = 0; j < 9; ++j) {
        float w = (invv[j] == 0) ? lg[j] : 0.f;
        _Float16* wp = &sm_w[idxv[j] * TILE + tid];  // column owned by this thread
        *wp = (_Float16)((float)(*wp) + w);          // duplicates accumulate
        atomicAdd(&sm_den[idxv[j]], w);
      }
      if (finalPass) {
#pragma unroll
        for (int j = 0; j < 9; ++j)
          out_assoc[((size_t)b * 9 + j) * HW_ + hw] = lg[j];
        int lab = idxv[jstar];
        g_labels[(size_t)b * HW_ + hw] = lab;
        out_lab[(size_t)b * HW_ + hw]  = (float)lab;
      }
    }
    __syncthreads();

    // ---- GEMM2: num[sp][c] += W[sp][px] * x[c][px], accumulate in LDS ----
    for (int t = wave; t < 128; t += 4) {
      const int m = t >> 3, n = t & 7;
      const _Float16* wrow = sm_w  + (m * 16 + Lm) * TILE;   // A row: sp, K=px
      const _Float16* xrow = sm_xh + (n * 16 + Lm) * TILE;   // B col: c,  K=px
      float* np = sm_num + (m * 16 + 8 * hs) * C_ + n * 16 + Lm;
      v8f acc;
#pragma unroll
      for (int r = 0; r < 8; ++r) acc[r] = np[r * C_];
      acc = __builtin_amdgcn_wmma_f32_16x16x32_f16(false, load_fragA(wrow, 0, hs),
                                                   false, load_fragB(xrow, 0, hs),
                                                   (short)0, acc, false, false);
      acc = __builtin_amdgcn_wmma_f32_16x16x32_f16(false, load_fragA(wrow, 32, hs),
                                                   false, load_fragB(xrow, 32, hs),
                                                   (short)0, acc, false, false);
#pragma unroll
      for (int r = 0; r < 8; ++r) np[r * C_] = acc[r];
    }
    __syncthreads();
  }

  // ---- single atomic flush of per-WG partials ----
  float* gnb = g_num + (size_t)b * K_ * C_;
  for (int i = tid; i < K_ * C_; i += 128) {
    float v = sm_num[i];
    if (v != 0.f) atomicAdd(&gnb[i], v);
  }
  for (int i = tid; i < K_; i += 128) {
    float v = sm_den[i];
    if (v != 0.f) atomicAdd(&g_den[b * K_ + i], v);
  }
}

// ---------------------------------------------------------------------------
// Reconstruction: recon[b,c,h,w] = feat[b, labels[b,h,w], c]
// ---------------------------------------------------------------------------
__global__ void __launch_bounds__(256)
recon_kernel(const float* __restrict__ feat_t, const int* __restrict__ labels,
             float* __restrict__ out) {
  size_t t = (size_t)blockIdx.x * 256 + threadIdx.x;  // over B_*C_*HW_
  if (t >= (size_t)B_ * C_ * HW_) return;
  int b = (int)(t / ((size_t)C_ * HW_));
  int rem = (int)(t % ((size_t)C_ * HW_));
  int c = rem / HW_, hw = rem % HW_;
  int lab = labels[(size_t)b * HW_ + hw];
  out[t] = feat_t[((size_t)(b * K_ + lab)) * C_ + c];
}

// ---------------------------------------------------------------------------
extern "C" void kernel_launch(void* const* d_in, const int* in_sizes, int n_in,
                              void* d_out, int out_size, void* d_ws, size_t ws_size,
                              hipStream_t stream) {
  (void)in_sizes; (void)n_in; (void)out_size; (void)ws_size;
  const float* x          = (const float*)d_in[0];
  const int*   p2sp       = (const int*)d_in[1];
  const int*   invis      = (const int*)d_in[2];
  const int*   init_index = (const int*)d_in[3];
  // d_in[4] = cir_index (unused by reference); d_in[5]/[6] = spixel_h/w (K=256 fixed)

  char* ws = (char*)d_ws;
  float*    g_num  = (float*)(ws);                 // 524288 B
  float*    g_den  = (float*)(ws + 524288);        //   4096 B (adjacent: joint memset)
  float*    feat_t = (float*)(ws + 528384);        // 524288 B
  _Float16* feat_h = (_Float16*)(ws + 1052672);    // 262144 B
  float*    s2     = (float*)(ws + 1314816);       //   4096 B
  int*      labels = (int*)(ws + 1318912);         // 1048576 B

  float* out_recon = (float*)d_out;                         // B*C*H*W
  float* out_lab   = out_recon + (size_t)B_ * C_ * HW_;     // B*H*W (labels as float)
  float* out_assoc = out_lab + (size_t)B_ * HW_;            // B*9*H*W

  hipFuncSetAttribute(reinterpret_cast<const void*>(&fused_iter_kernel),
                      hipFuncAttributeMaxDynamicSharedMemorySize, FUSED_LDS);
  hipFuncSetAttribute(reinterpret_cast<const void*>(&init_feat_kernel),
                      hipFuncAttributeMaxDynamicSharedMemorySize, INIT_LDS);

  hipMemsetAsync(g_num, 0, 528384, stream);
  init_feat_kernel<<<B_ * HW_ / 256, 256, INIT_LDS, stream>>>(x, init_index, g_num, g_den);

  for (int it = 0; it < NITER; ++it) {
    finalize_kernel<<<(B_ * K_ + 255) / 256, 256, 0, stream>>>(
        g_num, g_den, it == 0 ? 1.0f : EPS_, feat_t, feat_h, s2);
    hipMemsetAsync(g_num, 0, 528384, stream);
    fused_iter_kernel<<<B_ * NWG, 128, FUSED_LDS, stream>>>(
        x, p2sp, invis, feat_h, s2, g_num, g_den, it == NITER - 1 ? 1 : 0,
        out_assoc, out_lab, labels);
  }
  finalize_kernel<<<(B_ * K_ + 255) / 256, 256, 0, stream>>>(
      g_num, g_den, EPS_, feat_t, feat_h, s2);
  recon_kernel<<<((size_t)B_ * C_ * HW_ + 255) / 256, 256, 0, stream>>>(
      feat_t, labels, out_recon);
}
